// SSL_block_62182536511796
// MI455X (gfx1250) — matrix operand order repaired and checked by reading, not compiled, not run on hardware
//
#include <hip/hip_runtime.h>
#include <hip/hip_bf16.h>
#include <hip/hip_fp16.h>

typedef _Float16 v4h  __attribute__((ext_vector_type(4)));
typedef _Float16 v8h  __attribute__((ext_vector_type(8)));
typedef _Float16 v16h __attribute__((ext_vector_type(16)));
typedef float    v8f  __attribute__((ext_vector_type(8)));

#define B_N 8192
#define D_K 128
#define LOG2E 1.4426950408889634f

__device__ __forceinline__ float wave_reduce_sum32(float v) {
    v += __shfl_xor(v, 1, 32);
    v += __shfl_xor(v, 2, 32);
    v += __shfl_xor(v, 4, 32);
    v += __shfl_xor(v, 8, 32);
    v += __shfl_xor(v, 16, 32);
    return v;
}

// ---------------------------------------------------------------------------
// Kernel 0: zero tot_rating[B] and avg slot (d_out is poisoned by harness).
// ---------------------------------------------------------------------------
__global__ void init_tail_kernel(float* __restrict__ tail, int n) {
    int i = blockIdx.x * 256 + threadIdx.x;
    if (i < n) tail[i] = 0.0f;
}

// ---------------------------------------------------------------------------
// Kernel 1: per-row preprocessing. One wave (32 lanes) per output row.
//  - gather + l2-normalize x1/x2 rows (D=128, 4 floats/lane)
//  - pos[row] = dot(f1, f2)
//  - f16 copies of f1/f2 for the WMMA GEMM
//  - MLP: h=f1*f2 -> W1(128x128)+b1 -> leaky(0.2) -> W2+b2 -> sigmoid -> temp
//  - rowscale[row] = log2e / temp ; accumulate mean(temp)
// ---------------------------------------------------------------------------
__global__ void __launch_bounds__(256)
prep_kernel(const float* __restrict__ x1, const float* __restrict__ x2,
            const int* __restrict__ i1, const int* __restrict__ i2,
            const float* __restrict__ W1, const float* __restrict__ b1,
            const float* __restrict__ W2, const float* __restrict__ b2,
            _Float16* __restrict__ x1h, _Float16* __restrict__ x2h,
            float* __restrict__ rowscale, float* __restrict__ pos,
            float* __restrict__ avg_accum) {
    __shared__ float hsh[8][D_K];
    const int wave = threadIdx.x >> 5;
    const int lane = threadIdx.x & 31;
    const int row  = blockIdx.x * 8 + wave;

    const int idx1 = i1[row];
    const int idx2 = i2[row];
    const float4 v1 = ((const float4*)(x1 + (size_t)idx1 * D_K))[lane];
    const float4 v2 = ((const float4*)(x2 + (size_t)idx2 * D_K))[lane];

    float n1 = wave_reduce_sum32(v1.x*v1.x + v1.y*v1.y + v1.z*v1.z + v1.w*v1.w);
    float n2 = wave_reduce_sum32(v2.x*v2.x + v2.y*v2.y + v2.z*v2.z + v2.w*v2.w);
    const float inv1 = 1.0f / fmaxf(sqrtf(n1), 1e-12f);
    const float inv2 = 1.0f / fmaxf(sqrtf(n2), 1e-12f);

    const float f1x = v1.x*inv1, f1y = v1.y*inv1, f1z = v1.z*inv1, f1w = v1.w*inv1;
    const float f2x = v2.x*inv2, f2y = v2.y*inv2, f2z = v2.z*inv2, f2w = v2.w*inv2;
    const float hx = f1x*f2x, hy = f1y*f2y, hz = f1z*f2z, hw = f1w*f2w;

    const float posr = wave_reduce_sum32(hx + hy + hz + hw);

    // f16 feature copies for the WMMA GEMM (8-byte aligned stores)
    v4h p1; p1[0] = (_Float16)f1x; p1[1] = (_Float16)f1y; p1[2] = (_Float16)f1z; p1[3] = (_Float16)f1w;
    v4h p2; p2[0] = (_Float16)f2x; p2[1] = (_Float16)f2y; p2[2] = (_Float16)f2z; p2[3] = (_Float16)f2w;
    *(v4h*)(x1h + (size_t)row * D_K + lane * 4) = p1;
    *(v4h*)(x2h + (size_t)row * D_K + lane * 4) = p2;

    hsh[wave][lane*4 + 0] = hx;
    hsh[wave][lane*4 + 1] = hy;
    hsh[wave][lane*4 + 2] = hz;
    hsh[wave][lane*4 + 3] = hw;
    __syncthreads();

    // MLP: each lane computes 4 outputs o = lane + 32*c, streaming W1 rows.
    float z = 0.0f;
    #pragma unroll
    for (int c = 0; c < 4; ++c) {
        const int o = lane + 32 * c;
        const float4* wrow = (const float4*)(W1 + (size_t)o * D_K);
        float a = b1[o];
        #pragma unroll 8
        for (int kk = 0; kk < 32; ++kk) {
            const float4 w4 = wrow[kk];
            a += hsh[wave][kk*4+0] * w4.x + hsh[wave][kk*4+1] * w4.y +
                 hsh[wave][kk*4+2] * w4.z + hsh[wave][kk*4+3] * w4.w;
        }
        a = (a >= 0.0f) ? a : 0.2f * a;           // leaky_relu(0.2)
        z += a * W2[o];
    }
    z = wave_reduce_sum32(z) + b2[0];
    const float sgm  = 1.0f / (1.0f + __builtin_amdgcn_exp2f(-z * LOG2E)); // sigmoid
    const float temp = (1.0f - 0.05f) * (1.0f - sgm) + 0.05f;

    if (lane == 0) {
        rowscale[row] = LOG2E / temp;
        pos[row]      = posr;
        atomicAdd(avg_accum, temp * (1.0f / (float)B_N));
    }
}

// ---------------------------------------------------------------------------
// Kernel 2: fused WMMA GEMM + exp + row-sum + pos_rating store.
// Block = 128 threads (4 waves). Each wave owns row tile i0 (A resident in
// 32 VGPRs) and iterates 16 consecutive 16-wide column tiles.
// VGPR layouts per CDNA5 ISA:
//   A (16x32 f16): lane L=(lo,hi): a[0..7]=K(32kc+8hi+0..7), a[8..15]=+16
//   B (32x16 f16): lane L: col=lo, b[0..7]=K(32kc+8hi+0..7), b[8..15]=+16
//   C (16x16 f32): c[r] -> row r+8*hi, col lo
// ---------------------------------------------------------------------------
__global__ void __launch_bounds__(128)
gemm_exp_kernel(const _Float16* __restrict__ x1h, const _Float16* __restrict__ x2h,
                const float* __restrict__ rowscale, const float* __restrict__ pos,
                float* __restrict__ pos_rating, float* __restrict__ tot_rating) {
    const int lane = threadIdx.x & 31;
    const int wave = threadIdx.x >> 5;
    const int lo   = lane & 15;
    const int hi   = lane >> 4;
    const int i0   = blockIdx.y * 16;
    const int colwave = blockIdx.x * 4 + wave;    // 0..31, each does 16 j-tiles

    // Preload A tile for K=0..127 (kept in registers for all column tiles)
    const _Float16* Abase = x1h + (size_t)(i0 + lo) * D_K;
    v16h a[4];
    #pragma unroll
    for (int kc = 0; kc < 4; ++kc) {
        v8h alo = *(const v8h*)(Abase + 32*kc + 8*hi);
        v8h ahi = *(const v8h*)(Abase + 32*kc + 16 + 8*hi);
        a[kc] = __builtin_shufflevector(alo, ahi, 0,1,2,3,4,5,6,7,8,9,10,11,12,13,14,15);
    }

    float rs[8];
    #pragma unroll
    for (int r = 0; r < 8; ++r) rs[r] = rowscale[i0 + r + 8*hi];

    float sumrow[8] = {0.f,0.f,0.f,0.f,0.f,0.f,0.f,0.f};

    #pragma unroll 1
    for (int it = 0; it < 16; ++it) {
        const int j0 = (colwave * 16 + it) * 16;
        const _Float16* Bb = x2h + (size_t)(j0 + lo) * D_K;

        v8f c = {};
        #pragma unroll
        for (int kc = 0; kc < 4; ++kc) {
            v8h blo = *(const v8h*)(Bb + 32*kc + 8*hi);
            v8h bhi = *(const v8h*)(Bb + 32*kc + 16 + 8*hi);
            v16h b = __builtin_shufflevector(blo, bhi, 0,1,2,3,4,5,6,7,8,9,10,11,12,13,14,15);
            c = __builtin_amdgcn_wmma_f32_16x16x32_f16(
                    false, a[kc], false, b, (short)0, c, false, false);
        }

        const float posj = pos[j0 + lo];
        #pragma unroll
        for (int r = 0; r < 8; ++r) {
            const float sc = rs[r];
            sumrow[r] += __builtin_amdgcn_exp2f(c[r] * sc);         // exp(tot/temp)
            pos_rating[(size_t)(i0 + r + 8*hi) * B_N + j0 + lo] =
                __builtin_amdgcn_exp2f(posj * sc);                  // exp(pos[j]/temp[i])
        }
    }

    // Reduce row partial sums across the 16 column lanes of each half-wave.
    #pragma unroll
    for (int r = 0; r < 8; ++r) {
        float v = sumrow[r];
        v += __shfl_xor(v, 1, 32);
        v += __shfl_xor(v, 2, 32);
        v += __shfl_xor(v, 4, 32);
        v += __shfl_xor(v, 8, 32);
        if (lo == 0) atomicAdd(&tot_rating[i0 + r + 8*hi], v);
    }
}

// ---------------------------------------------------------------------------
extern "C" void kernel_launch(void* const* d_in, const int* in_sizes, int n_in,
                              void* d_out, int out_size, void* d_ws, size_t ws_size,
                              hipStream_t stream) {
    (void)in_sizes; (void)n_in; (void)out_size; (void)ws_size;
    const float* x1 = (const float*)d_in[0];
    const float* x2 = (const float*)d_in[1];
    const int*   i1 = (const int*)d_in[2];
    const int*   i2 = (const int*)d_in[3];
    const float* W1 = (const float*)d_in[4];
    const float* b1 = (const float*)d_in[5];
    const float* W2 = (const float*)d_in[6];
    const float* b2 = (const float*)d_in[7];

    float* out        = (float*)d_out;
    float* pos_rating = out;                          // [B,B]
    float* tot_rating = out + (size_t)B_N * B_N;      // [B]
    float* avg_slot   = tot_rating + B_N;             // scalar

    char* ws = (char*)d_ws;
    _Float16* x1h = (_Float16*)ws;  ws += (size_t)B_N * D_K * sizeof(_Float16);
    _Float16* x2h = (_Float16*)ws;  ws += (size_t)B_N * D_K * sizeof(_Float16);
    float* rowscale = (float*)ws;   ws += (size_t)B_N * sizeof(float);
    float* pos      = (float*)ws;

    init_tail_kernel<<<(B_N + 1 + 255) / 256, 256, 0, stream>>>(tot_rating, B_N + 1);
    prep_kernel<<<B_N / 8, 256, 0, stream>>>(x1, x2, i1, i2, W1, b1, W2, b2,
                                             x1h, x2h, rowscale, pos, avg_slot);
    gemm_exp_kernel<<<dim3(8, B_N / 16), 128, 0, stream>>>(x1h, x2h, rowscale, pos,
                                                           pos_rating, tot_rating);
}
